// RSSM_33938831573273
// MI455X (gfx1250) — matrix operand order repaired and checked by reading, not compiled, not run on hardware
//
#include <hip/hip_runtime.h>
#include <hip/hip_bf16.h>
#include <stdint.h>

// ---------------------------------------------------------------------------
// RSSM forward for MI455X (gfx1250, wave32, WMMA).
// 4 sequential scans, each: 16 cooperative workgroups x 256 threads (8 waves),
// grid barrier between phases, bf16 WMMA GEMMs with f32 accumulation.
// Weights (~25 MB bf16) are L2-resident (192 MB); the workload is latency-
// bound by the T=64 serial recurrence, so bf16 WMMA (K=32/instr) + LDS-staged
// activations is the right operating point. All global traffic goes through
// address_space(1) accessors so it lowers to global_* (not flat_*) ops.
// ---------------------------------------------------------------------------

#define NWG 16
#define TPB 256
#define LDS_PITCH 136   // 128 staged bf16 + 8 pad (272B row: avoids bank camping)

typedef unsigned short bf16_t;
typedef __attribute__((ext_vector_type(16))) __bf16 v16bf;
typedef __attribute__((ext_vector_type(8)))  float  v8f;
typedef __attribute__((ext_vector_type(4)))  int    v4i_;

union Frag { v4i_ q[2]; v16bf v; };

#if defined(__AMDGCN__) && __has_builtin(__builtin_amdgcn_global_load_async_to_lds_b128)
#define RSSM_ASYNC_LDS 1
typedef __attribute__((address_space(1))) v4i_ gv4i_t;   // global int4
typedef __attribute__((address_space(3))) v4i_ lv4i_t;   // LDS int4
#endif

// --- global (address_space(1)) accessors: force global_* instead of flat_* --
__device__ __forceinline__ v4i_ gload_b128(const void* p) {
  return *(const __attribute__((address_space(1))) v4i_*)p;
}
__device__ __forceinline__ float gload_f32(const float* p) {
  return *(const __attribute__((address_space(1))) float*)p;
}
__device__ __forceinline__ bf16_t gload_b16(const bf16_t* p) {
  return *(const __attribute__((address_space(1))) bf16_t*)p;
}
__device__ __forceinline__ void gstore_f32(float* p, float v) {
  *(__attribute__((address_space(1))) float*)p = v;
}
__device__ __forceinline__ void gstore_b16(bf16_t* p, bf16_t v) {
  *(__attribute__((address_space(1))) bf16_t*)p = v;
}

__device__ __forceinline__ bf16_t f2bf(float f) {
  unsigned u = __float_as_uint(f);
  u += 0x7fffu + ((u >> 16) & 1u);   // round-to-nearest-even
  return (bf16_t)(u >> 16);
}

__device__ __forceinline__ unsigned mix32(unsigned x) {
  x ^= x >> 16; x *= 0x7feb352dU;
  x ^= x >> 15; x *= 0x846ca68bU;
  x ^= x >> 16; return x;
}

__device__ __forceinline__ float sigmoidf_(float x) { return 1.f / (1.f + __expf(-x)); }
__device__ __forceinline__ float tanh_fast(float x) {
  x = fminf(fmaxf(x, -15.f), 15.f);
  float e = __expf(2.f * x);
  return (e - 1.f) / (e + 1.f);
}

// ---------------------------------------------------------------------------
// Grid-wide barrier across the NWG cooperative workgroups (generation based).
// Also emits the CDNA5 cluster split-barrier (S_NOP when ClusterID==0).
// ---------------------------------------------------------------------------
__device__ __forceinline__ void grid_barrier(int* cnt, int* gen) {
  __syncthreads();
#if __has_builtin(__builtin_amdgcn_s_cluster_barrier)
  __builtin_amdgcn_s_cluster_barrier();
#endif
  if (threadIdx.x == 0) {
    __threadfence();
    volatile int* vgen = (volatile int*)gen;
    int g  = *vgen;
    int my = atomicAdd(cnt, 1);
    if (my == NWG - 1) {
      *(volatile int*)cnt = 0;
      __threadfence();
      atomicAdd(gen, 1);
    } else {
      while (*vgen == g) __builtin_amdgcn_s_sleep(2);
    }
    __threadfence();
  }
  __syncthreads();
}

// ---------------------------------------------------------------------------
// Cooperative GEMM slab: OUT[128 x (NWG*NT*16)], this workgroup owns cols
// [blockIdx.x*NT*16, +NT*16). Wave w computes rows [16w,16w+16).
// A is bf16 [128][1024] row-major (A0 for k<1024, A1 for k>=1024).
// Wt is bf16 [Ntot][K] (pre-transposed weights: row = output col).
// A K-chunk of 128 is staged in LDS (async-to-LDS when available) and reused
// by all NT column tiles; B streams from L2-resident weights via global_*.
// ---------------------------------------------------------------------------
template <int NT>
__device__ void gemm_slab(const bf16_t* __restrict__ A0, const bf16_t* __restrict__ A1,
                          const bf16_t* __restrict__ Wt, int K,
                          const float* __restrict__ bias, int doElu, int addAct,
                          const float* __restrict__ act, const float* __restrict__ wact,
                          float* __restrict__ dstF, int pitchF,
                          float* __restrict__ dstF2, int pitchF2,
                          bf16_t* __restrict__ dstB, bf16_t* lds_a) {
  const int lane = threadIdx.x & 31;
  const int wid  = threadIdx.x >> 5;
  const int lm = lane & 15, lh = lane >> 4;
  const int mbase    = wid * 16;
  const int ncolbase = (int)blockIdx.x * (NT * 16);

  v8f acc[NT];
#pragma unroll
  for (int i = 0; i < NT; ++i) acc[i] = {};

  for (int kc = 0; kc < K; kc += 128) {
    const bf16_t* Asrc = (kc < 1024) ? (A0 + kc) : (A1 + (kc - 1024));
    __syncthreads();  // protect LDS slab reuse
    // stage A[0:128][kc:kc+128] -> LDS (2048 x 16B chunks / 256 threads)
    for (int c = (int)threadIdx.x; c < 2048; c += TPB) {
      const int row = c >> 4;
      const int ck  = (c & 15) * 8;  // 8 bf16 per 16B chunk
      const bf16_t* gp = Asrc + row * 1024 + ck;
      bf16_t* lp = lds_a + row * LDS_PITCH + ck;
#if RSSM_ASYNC_LDS
      __builtin_amdgcn_global_load_async_to_lds_b128(
          (gv4i_t*)gp, (lv4i_t*)lp, 0, 0);
#else
      *(v4i_*)lp = gload_b128(gp);
#endif
    }
#if RSSM_ASYNC_LDS
#if __has_builtin(__builtin_amdgcn_s_wait_asynccnt)
    __builtin_amdgcn_s_wait_asynccnt(0);
#else
    asm volatile("s_wait_asynccnt 0" ::: "memory");
#endif
#endif
    __syncthreads();

    if (kc + 128 < K)
      __builtin_prefetch(Wt + (size_t)(ncolbase + lm) * K + kc + 128, 0, 1);

#pragma unroll
    for (int k8 = 0; k8 < 4; ++k8) {
      const int k0 = k8 * 32;
      // A fragment (contiguous K order, matching B):
      // lane<16: K = k0+0..15 ; lane>=16: K = k0+16..31
      const bf16_t* ap = lds_a + (mbase + lm) * LDS_PITCH + k0 + lh * 16;
      Frag a;
      a.q[0] = *(const v4i_*)ap;          // ds_load_b128
      a.q[1] = *(const v4i_*)(ap + 8);
#pragma unroll
      for (int nt = 0; nt < NT; ++nt) {
        const int ncol = ncolbase + nt * 16 + lm;
        // B fragment: column = lane&15, 16 contiguous K (lane half selects K±16)
        const bf16_t* bp = Wt + (size_t)ncol * K + kc + k0 + lh * 16;
        Frag b;
        b.q[0] = gload_b128(bp);          // global_load_b128
        b.q[1] = gload_b128(bp + 8);
        acc[nt] = __builtin_amdgcn_wmma_f32_16x16x32_bf16(
            false, a.v, false, b.v, (short)0, acc[nt], false, false);
      }
    }
  }

  // Epilogue: D layout — VGPR i maps to row mbase+i (+8 for lanes 16..31)
#pragma unroll
  for (int nt = 0; nt < NT; ++nt) {
    const int ncol = ncolbase + nt * 16 + lm;
    const float bv = bias ? gload_f32(bias + ncol) : 0.f;
#pragma unroll
    for (int i = 0; i < 8; ++i) {
      const int row = mbase + i + lh * 8;
      float v = acc[nt][i] + bv;
      if (addAct) {  // action tail of w_pre (rows 1024..1029), kept fp32
#pragma unroll
        for (int a6 = 0; a6 < 6; ++a6)
          v += gload_f32(act + row * 6 + a6) * gload_f32(wact + a6 * 1024 + ncol);
      }
      if (doElu) v = v > 0.f ? v : (__expf(v) - 1.f);
      if (dstF)  gstore_f32(dstF  + (size_t)row * pitchF  + ncol, v);
      if (dstF2) gstore_f32(dstF2 + (size_t)row * pitchF2 + ncol, v);
      if (dstB)  gstore_b16(dstB + row * 1024 + ncol, f2bf(v));
    }
  }
}

// ---------------------------------------------------------------------------
// LayerNorm over 3072 + GRU gates. One wave per batch row (128 waves total).
// ---------------------------------------------------------------------------
__device__ void ln_gru_phase(const float* __restrict__ parts,
                             const float* __restrict__ lnS, const float* __restrict__ lnO,
                             float* __restrict__ deterF, bf16_t* __restrict__ deterB,
                             float* __restrict__ feat) {
  const int lane = threadIdx.x & 31;
  const int row  = (int)blockIdx.x * 8 + (int)(threadIdx.x >> 5);
  const float* p = parts + (size_t)row * 3072;
  float s = 0.f, ss = 0.f;
  for (int j = lane; j < 3072; j += 32) { float x = gload_f32(p + j); s += x; ss += x * x; }
#pragma unroll
  for (int off = 16; off > 0; off >>= 1) {
    s  += __shfl_xor(s, off, 32);
    ss += __shfl_xor(ss, off, 32);
  }
  const float mean = s * (1.f / 3072.f);
  const float var  = fmaxf(ss * (1.f / 3072.f) - mean * mean, 0.f);
  const float rstd = rsqrtf(var + 1e-5f);
  for (int col = lane; col < 1024; col += 32) {
    float rn = (gload_f32(p + col)        - mean) * rstd * gload_f32(lnS + col)        + gload_f32(lnO + col);
    float cn = (gload_f32(p + col + 1024) - mean) * rstd * gload_f32(lnS + col + 1024) + gload_f32(lnO + col + 1024);
    float un = (gload_f32(p + col + 2048) - mean) * rstd * gload_f32(lnS + col + 2048) + gload_f32(lnO + col + 2048);
    float r = sigmoidf_(rn);
    float c = tanh_fast(r * cn);
    float u = sigmoidf_(un - 1.f);
    float h = gload_f32(deterF + row * 1024 + col);
    float nd = u * c + (1.f - u) * h;
    gstore_f32(deterF + row * 1024 + col, nd);
    gstore_b16(deterB + row * 1024 + col, f2bf(nd));
    if (feat) gstore_f32(feat + (size_t)row * 2048 + col, nd);
  }
}

// ---------------------------------------------------------------------------
// Straight-through categorical sample: forward value is exactly the one-hot
// argmax(logits + gumbel). Deterministic hash-based gumbel (stand-in PRNG).
// One thread per (batch, s-block): 16 WGs * 256 threads = 4096 = 128*32.
// ---------------------------------------------------------------------------
__device__ void sample_phase(const float* __restrict__ stats, unsigned key,
                             bf16_t* __restrict__ stochB, float* __restrict__ feat) {
  const int gt = (int)(blockIdx.x * TPB + threadIdx.x);
  const int b = gt >> 5, s5 = gt & 31;
  const float* lg = stats + (size_t)b * 1024 + s5 * 32;
  const unsigned seed = key * 0x9E3779B9u + 0x85EBCA6Bu;
  int best = 0; float bestv = -3.4e38f;
  for (int d = 0; d < 32; ++d) {
    unsigned h = mix32(seed ^ ((unsigned)(gt * 32 + d) * 0x27D4EB2Du));
    float u = (float)(h >> 8) * (1.f / 16777216.f) + (0.5f / 16777216.f);
    float g = -__logf(-__logf(u));
    float v = gload_f32(lg + d) + g;
    if (v > bestv) { bestv = v; best = d; }
  }
  for (int d = 0; d < 32; ++d) {
    gstore_b16(stochB + (size_t)b * 1024 + s5 * 32 + d,
               (d == best) ? (bf16_t)0x3F80 : (bf16_t)0);
    if (feat) gstore_f32(feat + (size_t)b * 2048 + 1024 + s5 * 32 + d,
                         (d == best) ? 1.f : 0.f);
  }
}

// ---------------------------------------------------------------------------
struct ScanArgs {
  const float*  actions;   // [64][128][6]
  const bf16_t* embbf;     // [64][128][1024]
  const bf16_t* WtPre; const bf16_t* WtGru; const bf16_t* Wt1; const bf16_t* Wt2;
  const float*  bPre;  const float*  bGru;  const float*  b1;  const float*  b2;
  const float*  lnS;   const float*  lnO;   const float*  wact;
  bf16_t* stochB; bf16_t* deterB; bf16_t* Xbf;
  float*  deterF; float*  parts;  float*  statsWs;
  float*  statsOut;   // nullable: [64][128][1024]
  float*  featOut;    // nullable: [64][128][2048]
  int* bcnt; int* bgen;
  int mode;           // 0 = prior head, 1 = posterior head
  unsigned keyAdd;
};

__global__ __launch_bounds__(TPB, 1) void rssm_scan(ScanArgs p) {
  __shared__ __align__(16) bf16_t lds_a[128 * LDS_PITCH];

  // zero initial state (stoch = 0, deter = 0)
  for (int i = (int)(blockIdx.x * TPB + threadIdx.x); i < 128 * 1024; i += NWG * TPB) {
    gstore_b16(p.stochB + i, 0);
    gstore_b16(p.deterB + i, 0);
    gstore_f32(p.deterF + i, 0.f);
  }
  grid_barrier(p.bcnt, p.bgen);

  for (int t = 0; t < 64; ++t) {
    const float* act_t  = p.actions + (size_t)t * 128 * 6;
    float* feat_t  = p.featOut  ? p.featOut  + (size_t)t * 128 * 2048 : nullptr;
    float* stats_t = p.statsOut ? p.statsOut + (size_t)t * 128 * 1024 : nullptr;

    // X = elu([stoch, act] @ w_pre + b_pre)           [128,1030]x[1030,1024]
    gemm_slab<4>(p.stochB, nullptr, p.WtPre, 1024, p.bPre, 1, 1, act_t, p.wact,
                 nullptr, 0, nullptr, 0, p.Xbf, lds_a);
    grid_barrier(p.bcnt, p.bgen);

    // parts = [X, h] @ w_gru + b_gru                  [128,2048]x[2048,3072]
    gemm_slab<12>(p.Xbf, p.deterB, p.WtGru, 2048, p.bGru, 0, 0, nullptr, nullptr,
                  p.parts, 3072, nullptr, 0, nullptr, lds_a);
    grid_barrier(p.bcnt, p.bgen);

    // LayerNorm + GRU gates -> new deter (also feature[:,0:1024])
    ln_gru_phase(p.parts, p.lnS, p.lnO, p.deterF, p.deterB, feat_t);
    grid_barrier(p.bcnt, p.bgen);

    // head: Y = elu(A @ W1 + b1)
    if (p.mode == 0) {  // prior: A = nd               [128,1024]x[1024,1024]
      gemm_slab<4>(p.deterB, nullptr, p.Wt1, 1024, p.b1, 1, 0, nullptr, nullptr,
                   nullptr, 0, nullptr, 0, p.Xbf, lds_a);
    } else {            // post:  A = [nd, emb_t]      [128,2048]x[2048,1024]
      gemm_slab<4>(p.deterB, p.embbf + (size_t)t * 128 * 1024, p.Wt1, 2048, p.b1,
                   1, 0, nullptr, nullptr, nullptr, 0, nullptr, 0, p.Xbf, lds_a);
    }
    grid_barrier(p.bcnt, p.bgen);

    // stats = Y @ W2 + b2                             [128,1024]x[1024,1024]
    gemm_slab<4>(p.Xbf, nullptr, p.Wt2, 1024, p.b2, 0, 0, nullptr, nullptr,
                 p.statsWs, 1024, stats_t, 1024, nullptr, lds_a);
    grid_barrier(p.bcnt, p.bgen);

    // categorical straight-through sample -> next stoch (and feature[:,1024:])
    sample_phase(p.statsWs, p.keyAdd + (unsigned)t, p.stochB, feat_t);
    grid_barrier(p.bcnt, p.bgen);
  }
}

// ---------------------------------------------------------------------------
// One-time (per launch) weight prep: fp32 [K][N] -> bf16 [N][K] (transposed,
// WMMA B-fragment friendly: 32B contiguous K per output column).
// ---------------------------------------------------------------------------
__global__ void k_transpose_bf16(const float* __restrict__ src, bf16_t* __restrict__ dst,
                                 int K, int N) {
  size_t i = (size_t)blockIdx.x * blockDim.x + threadIdx.x;
  if (i >= (size_t)K * N) return;
  size_t k = i / (size_t)N, n = i % (size_t)N;
  gstore_b16(dst + n * (size_t)K + k, f2bf(gload_f32(src + i)));
}

__global__ void k_convert_bf16(const float* __restrict__ src, bf16_t* __restrict__ dst,
                               size_t n) {
  size_t i = (size_t)blockIdx.x * blockDim.x + threadIdx.x;
  if (i < n) gstore_b16(dst + i, f2bf(gload_f32(src + i)));
}

// ---------------------------------------------------------------------------
extern "C" void kernel_launch(void* const* d_in, const int* in_sizes, int n_in,
                              void* d_out, int out_size, void* d_ws, size_t ws_size,
                              hipStream_t stream) {
  (void)in_sizes; (void)n_in; (void)out_size; (void)ws_size;
  const float* embeds  = (const float*)d_in[0];
  const float* actions = (const float*)d_in[1];
  const float* w_pre   = (const float*)d_in[2];
  const float* b_pre   = (const float*)d_in[3];
  const float* w_gru   = (const float*)d_in[4];
  const float* b_gru   = (const float*)d_in[5];
  const float* ln_s    = (const float*)d_in[6];
  const float* ln_o    = (const float*)d_in[7];
  const float* w_p1    = (const float*)d_in[8];
  const float* b_p1    = (const float*)d_in[9];
  const float* w_p2    = (const float*)d_in[10];
  const float* b_p2    = (const float*)d_in[11];
  const float* w_q1    = (const float*)d_in[12];
  const float* b_q1    = (const float*)d_in[13];
  const float* w_q2    = (const float*)d_in[14];
  const float* b_q2    = (const float*)d_in[15];
  float* out = (float*)d_out;
  char*  ws  = (char*)d_ws;

  size_t o = 0;
  auto take = [&](size_t bytes) { size_t r = o; o += (bytes + 255) & ~(size_t)255; return r; };
  int*    bar    = (int*)   (ws + take(256));
  bf16_t* WtPre  = (bf16_t*)(ws + take((size_t)1024 * 1024 * 2));
  bf16_t* WtGru  = (bf16_t*)(ws + take((size_t)3072 * 2048 * 2));
  bf16_t* WtP1   = (bf16_t*)(ws + take((size_t)1024 * 1024 * 2));
  bf16_t* WtP2   = (bf16_t*)(ws + take((size_t)1024 * 1024 * 2));
  bf16_t* WtQ1   = (bf16_t*)(ws + take((size_t)1024 * 2048 * 2));
  bf16_t* WtQ2   = (bf16_t*)(ws + take((size_t)1024 * 1024 * 2));
  bf16_t* embbf  = (bf16_t*)(ws + take((size_t)64 * 128 * 1024 * 2));
  bf16_t* stochB = (bf16_t*)(ws + take((size_t)128 * 1024 * 2));
  bf16_t* deterB = (bf16_t*)(ws + take((size_t)128 * 1024 * 2));
  bf16_t* Xbf    = (bf16_t*)(ws + take((size_t)128 * 1024 * 2));
  float*  deterF = (float*) (ws + take((size_t)128 * 1024 * 4));
  float*  parts  = (float*) (ws + take((size_t)128 * 3072 * 4));
  float*  statsW = (float*) (ws + take((size_t)128 * 1024 * 4));

  (void)hipMemsetAsync(bar, 0, 256, stream);  // deterministic barrier state

  dim3 blk(TPB);
  auto grid1 = [](size_t n) { return dim3((unsigned)((n + TPB - 1) / TPB)); };

  k_transpose_bf16<<<grid1((size_t)1024 * 1024), blk, 0, stream>>>(w_pre, WtPre, 1024, 1024);
  k_transpose_bf16<<<grid1((size_t)2048 * 3072), blk, 0, stream>>>(w_gru, WtGru, 2048, 3072);
  k_transpose_bf16<<<grid1((size_t)1024 * 1024), blk, 0, stream>>>(w_p1, WtP1, 1024, 1024);
  k_transpose_bf16<<<grid1((size_t)1024 * 1024), blk, 0, stream>>>(w_p2, WtP2, 1024, 1024);
  k_transpose_bf16<<<grid1((size_t)2048 * 1024), blk, 0, stream>>>(w_q1, WtQ1, 2048, 1024);
  k_transpose_bf16<<<grid1((size_t)1024 * 1024), blk, 0, stream>>>(w_q2, WtQ2, 1024, 1024);
  k_convert_bf16<<<grid1((size_t)64 * 128 * 1024), blk, 0, stream>>>(
      embeds, embbf, (size_t)64 * 128 * 1024);

  ScanArgs a{};
  a.actions = actions; a.embbf = embbf;
  a.WtPre = WtPre; a.WtGru = WtGru;
  a.bPre = b_pre; a.bGru = b_gru;
  a.lnS = ln_s; a.lnO = ln_o;
  a.wact = w_pre + (size_t)1024 * 1024;   // w_pre rows 1024..1029 (action block)
  a.stochB = stochB; a.deterB = deterB; a.Xbf = Xbf;
  a.deterF = deterF; a.parts = parts; a.statsWs = statsW;
  a.bcnt = bar; a.bgen = bar + 1;

  float* priors = out;
  float* posts  = out + (size_t)64 * 128 * 1024;
  float* postF  = out + (size_t)2 * 64 * 128 * 1024;
  float* priorF = out + (size_t)2 * 64 * 128 * 1024 + (size_t)64 * 128 * 2048;

  // scan 1: prior stats
  a.mode = 0; a.Wt1 = WtP1; a.Wt2 = WtP2; a.b1 = b_p1; a.b2 = b_p2;
  a.statsOut = priors; a.featOut = nullptr; a.keyAdd = 0u;
  rssm_scan<<<dim3(NWG), blk, 0, stream>>>(a);
  // scan 2: posterior stats
  a.mode = 1; a.Wt1 = WtQ1; a.Wt2 = WtQ2; a.b1 = b_q1; a.b2 = b_q2;
  a.statsOut = posts; a.featOut = nullptr; a.keyAdd = 128u;
  rssm_scan<<<dim3(NWG), blk, 0, stream>>>(a);
  // scan 3: posterior features
  a.statsOut = nullptr; a.featOut = postF; a.keyAdd = 256u;
  rssm_scan<<<dim3(NWG), blk, 0, stream>>>(a);
  // scan 4: prior features
  a.mode = 0; a.Wt1 = WtP1; a.Wt2 = WtP2; a.b1 = b_p1; a.b2 = b_p2;
  a.statsOut = nullptr; a.featOut = priorF; a.keyAdd = 320u;
  rssm_scan<<<dim3(NWG), blk, 0, stream>>>(a);
}
